// Cyborg_46325517255370
// MI455X (gfx1250) — compile-verified, matrix operand order checked by reading, not run on hardware
//
#include <hip/hip_runtime.h>

// ---- constants from the reference ----
#define BB 256
#define SS 128
#define VV 512
#define ZO 1e-6f

typedef __attribute__((ext_vector_type(16))) _Float16 v16h;
typedef __attribute__((ext_vector_type(8)))  _Float16 v8h;
typedef __attribute__((ext_vector_type(8)))  float    v8f;

// Fragment element -> K mapping for 16x16x32 f16 WMMA A/B tiles (K=32):
//   k = (e/8)*16 + (lane/16)*8 + (e&7)  -> two contiguous 8-element runs
__device__ __forceinline__ int frag_k(int lane, int e) {
  return ((e >> 3) << 4) + ((lane >> 4) << 3) + (e & 7);
}

// Concatenated gate-column mapping.
// cols 0-3: pop[0], 4-7: pushop[0], 8-11: pop[1], 12-15: pushop[1],
// 16-79: pushfn[0], 80-143: pushfn[1], 144-207: calc
__device__ __forceinline__ const float* gate_row(int col,
    const float* pop_w, const float* pushop_w,
    const float* pushfn_w, const float* calc_w) {
  if (col < 4)   return pop_w    + col * 1536;
  if (col < 8)   return pushop_w + (col - 4) * 1536;
  if (col < 12)  return pop_w    + (4 + col - 8) * 1536;
  if (col < 16)  return pushop_w + (4 + col - 12) * 1536;
  if (col < 80)  return pushfn_w + (col - 16) * 1536;
  if (col < 144) return pushfn_w + (64 + col - 80) * 1536;
  return calc_w + (col - 144) * 1536;
}

__device__ __forceinline__ const float* gate_nw_row(int col,
    const float* pop_nw, const float* pushop_nw,
    const float* pushfn_nw, const float* calc_nw) {
  if (col < 4)   return pop_nw    + col * 3;
  if (col < 8)   return pushop_nw + (col - 4) * 3;
  if (col < 12)  return pop_nw    + (4 + col - 8) * 3;
  if (col < 16)  return pushop_nw + (4 + col - 12) * 3;
  if (col < 80)  return pushfn_nw + (col - 16) * 3;
  if (col < 144) return pushfn_nw + (64 + col - 80) * 3;
  return calc_nw + (col - 144) * 3;
}

// ---------------- prep kernels (weights are step-invariant) ----------------

__global__ void prep_norms(const float* __restrict__ pop_w,
                           const float* __restrict__ pushop_w,
                           const float* __restrict__ pushfn_w,
                           const float* __restrict__ calc_w,
                           float* __restrict__ invw) {
  __shared__ float red[256];
  int col = blockIdx.x / 3, j = blockIdx.x % 3;
  const float* row = gate_row(col, pop_w, pushop_w, pushfn_w, calc_w) + j * VV;
  float ss = 0.f;
  for (int v = threadIdx.x; v < VV; v += 256) { float f = row[v]; ss += f * f; }
  red[threadIdx.x] = ss; __syncthreads();
  for (int s = 128; s > 0; s >>= 1) {
    if (threadIdx.x < s) red[threadIdx.x] += red[threadIdx.x + s];
    __syncthreads();
  }
  if (threadIdx.x == 0) invw[col * 3 + j] = 1.f / fmaxf(sqrtf(red[0]), 1e-8f);
}

// normalized f16 weights pre-swizzled into WMMA B-fragment layout
// W1: [9 nt][3 j][16 kt][32 lane][16 e], W2: [4 nt][3 j][16 kt][32 lane][16 e]
__global__ void prep_wswz(const float* __restrict__ pop_w,
                          const float* __restrict__ pushop_w,
                          const float* __restrict__ pushfn_w,
                          const float* __restrict__ calc_w,
                          const float* __restrict__ invw,
                          _Float16* __restrict__ W1, _Float16* __restrict__ W2) {
  int idx = blockIdx.x * 256 + threadIdx.x;
  const int total1 = 9 * 3 * 16 * 512;
  const int total2 = 4 * 3 * 16 * 512;
  _Float16* dst; int li, colbase;
  if (idx < total1) { dst = W1 + idx; li = idx; colbase = 0; }
  else {
    int i2 = idx - total1;
    if (i2 >= total2) return;
    dst = W2 + i2; li = i2; colbase = 144;
  }
  int e = li & 15, lane = (li >> 4) & 31, kt = (li >> 9) & 15;
  int jn = li >> 13, j = jn % 3, nt = jn / 3;
  int col = colbase + nt * 16 + (lane & 15);
  int k = kt * 32 + frag_k(lane, e);
  const float* row = gate_row(col, pop_w, pushop_w, pushfn_w, calc_w);
  *dst = (_Float16)(row[j * VV + k] * invw[col * 3 + j]);
}

__global__ void prep_nw(const float* __restrict__ pop_nw,
                        const float* __restrict__ pushop_nw,
                        const float* __restrict__ pushfn_nw,
                        const float* __restrict__ calc_nw,
                        float* __restrict__ nw1, float* __restrict__ nw2) {
  int idx = blockIdx.x * 256 + threadIdx.x;
  if (idx >= 208 * 3) return;
  int col = idx / 3, j = idx % 3;
  const float* r = gate_nw_row(col, pop_nw, pushop_nw, pushfn_nw, calc_nw);
  float s = 1.f / (1.f + __expf(-r[j]));
  if (col < 144) nw1[col * 3 + j] = s;
  else           nw2[(col - 144) * 3 + j] = s;
}

// syms as B-fragments for the [16x32]x[32x16] output mixes (K=32: 30 syms + pad)
__global__ void prep_syms(const float* __restrict__ syms,
                          _Float16* __restrict__ symsSwz) {
  int idx = blockIdx.x * 256 + threadIdx.x;
  if (idx >= 32 * 32 * 16) return;
  int e = idx & 15, lane = (idx >> 4) & 31, vt = idx >> 9;
  int n = lane & 15;
  int k = frag_k(lane, e);
  float v = (k < 30) ? syms[k * VV + vt * 16 + n] : 0.f;
  symsSwz[idx] = (_Float16)v;
}

// ---------------- main recurrent kernel ----------------
// 16 workgroups x 16 batch rows; all stack state LDS-resident across 128 steps.
__global__ __launch_bounds__(256, 1) void cyborg_main(
    const float* __restrict__ x, const float* __restrict__ siv,
    const float* __restrict__ sharp,
    const float* __restrict__ nw1, const float* __restrict__ nw2,
    const _Float16* __restrict__ Wswz1, const _Float16* __restrict__ Wswz2,
    const _Float16* __restrict__ symsSwz,
    float* __restrict__ out) {
  extern __shared__ char smem[];
  // halves first (all 16B-aligned bases)
  _Float16* s_stacks = (_Float16*)smem;            // [2][16][4][512]   65536 h
  _Float16* s_inps   = s_stacks + 65536;           // [16][1536]        24576 h
  _Float16* s_aswz   = s_inps + 24576;             // [3][16][32][16]   24576 h
  _Float16* s_pval   = s_aswz + 24576;             // [2][16][512]      16384 h
  _Float16* s_cswz   = s_pval + 16384;             // [2][32][16]       1024 h
  // floats after
  float*    s_gates  = (float*)(s_cswz + 1024);    // [16][144]         2304 f
  float*    s_nacc4  = s_gates + 2304;             // [48][4]           192 f
  float*    s_coef   = s_nacc4 + 192;              // [2][16][32]       1024 f
  float*    s_upd    = s_coef + 1024;              // [2][16][4][3]     384 f
  float*    s_ptr    = s_upd + 384;                // [2][16][4]        128 f
  float*    s_ptr2   = s_ptr + 128;                // [2][16][4]        128 f

  const int tid = threadIdx.x;
  const int lane = tid & 31, wv = tid >> 5;
  const int m0 = blockIdx.x * 16;

  // ---- init: stacks = zero_offset, hard-push init vec into slot 1 ----
  for (int i = tid; i < 2 * 16 * 4 * 512; i += 256) {
    int v = i & 511, d = (i >> 9) & 3;
    s_stacks[i] = (_Float16)((d == 1) ? siv[v] : ZO);
  }
  for (int i = tid; i < 2 * 16 * 4; i += 256)
    s_ptr[i] = ((i & 3) == 1) ? 1.f : 0.f;
  __syncthreads();

#pragma unroll 1
  for (int t = 0; t < SS; t++) {
    // ---- step 1: load x_t (vectorized) + peeks (vectorized) ----
    for (int i = tid; i < 16 * 64; i += 256) {
      int c = i & 63, m = i >> 6;
      const float* px = &x[((size_t)(m0 + m) * SS + t) * VV + c * 8];
      float4 a = *(const float4*)px;
      float4 b = *(const float4*)(px + 4);
      v8h o;
      o[0] = (_Float16)a.x; o[1] = (_Float16)a.y;
      o[2] = (_Float16)a.z; o[3] = (_Float16)a.w;
      o[4] = (_Float16)b.x; o[5] = (_Float16)b.y;
      o[6] = (_Float16)b.z; o[7] = (_Float16)b.w;
      *(v8h*)(s_inps + m * 1536 + c * 8) = o;
      if (t + 1 < SS) __builtin_prefetch(px + VV, 0, 1);
    }
    for (int i = tid; i < 2 * 16 * 64; i += 256) {
      int c = i & 63, m = (i >> 6) & 15, k = i >> 10;
      const float* pp = s_ptr + (k * 16 + m) * 4;
      const v8h* st = (const v8h*)(s_stacks + ((k * 16 + m) * 4) * 512) + c;
      v8h s0 = st[0], s1 = st[64], s2 = st[128], s3 = st[192];
      v8h o;
#pragma unroll
      for (int e = 0; e < 8; e++)
        o[e] = (_Float16)((float)s0[e] * pp[0] + (float)s1[e] * pp[1] +
                          (float)s2[e] * pp[2] + (float)s3[e] * pp[3]);
      *(v8h*)(s_inps + m * 1536 + 512 + k * 512 + c * 8) = o;
    }
    __syncthreads();

    // ---- step 2: parallel norm partials (all 3 vecs), then A-fragments ----
    if (tid < 192) {  // 48 (m,j) groups x 4 partials, 128 halves each
      int g = tid >> 2, part = tid & 3;
      int m = g & 15, j = g >> 4;
      const v8h* q = (const v8h*)(s_inps + m * 1536 + j * 512 + part * 128);
      float ss = 0.f;
#pragma unroll
      for (int c = 0; c < 16; c++) {
        v8h h = q[c];
#pragma unroll
        for (int e = 0; e < 8; e++) { float f = (float)h[e]; ss += f * f; }
      }
      s_nacc4[tid] = ss;
    }
    __syncthreads();
    for (int i = tid; i < 3 * 16 * 32; i += 256) {
      int l = i & 31, kt = (i >> 5) & 15, j = i >> 9;
      int m = l & 15, kh = l >> 4;
      int g = j * 16 + m;
      float ss = s_nacc4[g * 4] + s_nacc4[g * 4 + 1] +
                 s_nacc4[g * 4 + 2] + s_nacc4[g * 4 + 3];
      float invn = 1.f / fmaxf(sqrtf(ss), 1e-8f);
      const v8h* q = (const v8h*)(s_inps + m * 1536 + j * 512 + kt * 32 + kh * 8);
      v8h c0 = q[0], c1 = q[2];  // two contiguous 8-half runs of the fragment
      v8h o0, o1;
#pragma unroll
      for (int e = 0; e < 8; e++) {
        o0[e] = (_Float16)((float)c0[e] * invn);
        o1[e] = (_Float16)((float)c1[e] * invn);
      }
      v8h* dst = (v8h*)(s_aswz + ((j * 16 + kt) * 32 + l) * 16);
      dst[0] = o0; dst[1] = o1;
    }
    __syncthreads();

    // ---- step 3: phase-1 gates (144 cols): WMMA + fused NAND product ----
    for (int nt = wv; nt < 9; nt += 8) {
      float prod[8];
#pragma unroll
      for (int r = 0; r < 8; r++) prod[r] = 1.f;
      int col = nt * 16 + (lane & 15);
#pragma unroll
      for (int j = 0; j < 3; j++) {
        v8f acc = {};
#pragma unroll
        for (int kt = 0; kt < 16; kt++) {
          v16h a = *(const v16h*)(s_aswz + ((j * 16 + kt) * 32 + lane) * 16);
          v16h b = *(const v16h*)(Wswz1 + (((nt * 3 + j) * 16 + kt) * 32 + lane) * 16);
          acc = __builtin_amdgcn_wmma_f32_16x16x32_f16(
              false, a, false, b, (short)0, acc, false, false);
        }
        float nws = nw1[col * 3 + j];
#pragma unroll
        for (int r = 0; r < 8; r++) {
          float c = fabsf(acc[r]);
          prod[r] *= nws * c + (1.f - nws) * (1.f - c);
        }
      }
#pragma unroll
      for (int r = 0; r < 8; r++)
        s_gates[(r + ((lane >> 4) << 3)) * 144 + col] = prod[r];
    }
    __syncthreads();

    // ---- step 4: pop/push probs, ptr roll, pushfn coefs, coef A-frags ----
    if (tid < 32) {
      int m = tid & 15, k = tid >> 4;
      const float* g = s_gates + m * 144 + k * 8;
      float sh = sharp[k];
      float prp = 1.f / (1.f + __expf((fmaxf(g[2], g[3]) - fmaxf(g[0], g[1])) * sh));
      float prn = 1.f - prp;
      float qp  = 1.f / (1.f + __expf((fmaxf(g[6], g[7]) - fmaxf(g[4], g[5])) * sh));
      float qn  = 1.f - qp;
      float* P = s_ptr + (k * 16 + m) * 4;
      float p1[4], pp[4];
#pragma unroll
      for (int d = 0; d < 4; d++) p1[d] = prp * P[(d + 1) & 3] + prn * P[d];
#pragma unroll
      for (int d = 0; d < 4; d++) pp[d] = p1[(d + 3) & 3];
      float* P2 = s_ptr2 + (k * 16 + m) * 4;
      float* U = s_upd + (k * 16 + m) * 12;
#pragma unroll
      for (int d = 0; d < 4; d++) {
        P2[d] = qp * pp[d] + qn * p1[d];
        float A = 1.f - prp * P[d], Bv = ZO * prp * P[d], C = qp * pp[d];
        U[d * 3 + 0] = A * (1.f - C);
        U[d * 3 + 1] = Bv * (1.f - C);
        U[d * 3 + 2] = C;
      }
    }
    for (int i = tid; i < 1024; i += 256) {  // pushfn coefs summed over R
      int c = i & 31, m = (i >> 5) & 15, k = i >> 9;
      int base = 16 + k * 64 + c * 2;
      s_coef[(k * 16 + m) * 32 + c] =
          s_gates[m * 144 + base] + s_gates[m * 144 + base + 1];
    }
    __syncthreads();
    if (tid < 64) {  // coef A-fragments (two contiguous runs)
      int l = tid & 31, k = tid >> 5, m = l & 15, kh = l >> 4;
      const float* cf = s_coef + (k * 16 + m) * 32;
      v8h o0, o1;
#pragma unroll
      for (int e = 0; e < 8; e++) {
        o0[e] = (_Float16)cf[kh * 8 + e];
        o1[e] = (_Float16)cf[16 + kh * 8 + e];
      }
      v8h* dst = (v8h*)(s_cswz + (k * 32 + l) * 16);
      dst[0] = o0; dst[1] = o1;
    }
    __syncthreads();

    // ---- step 5a: syms-mix WMMA -> s_pval ----
    for (int job = wv; job < 64; job += 8) {
      int k = job >> 5, vt = job & 31;
      v8f acc = {};
      v16h a = *(const v16h*)(s_cswz + (k * 32 + lane) * 16);
      v16h b = *(const v16h*)(symsSwz + (vt * 32 + lane) * 16);
      acc = __builtin_amdgcn_wmma_f32_16x16x32_f16(
          false, a, false, b, (short)0, acc, false, false);
      int v = vt * 16 + (lane & 15);
#pragma unroll
      for (int r = 0; r < 8; r++)
        s_pval[k * 8192 + (r + ((lane >> 4) << 3)) * 512 + v] = (_Float16)acc[r];
    }
    __syncthreads();
    // ---- step 5b: vectorized pop/push stack update ----
    for (int i = tid; i < 2 * 16 * 64; i += 256) {
      int c = i & 63, m = (i >> 6) & 15, k = i >> 10;
      const float* cf = s_coef + (k * 16 + m) * 32;
      const float* U = s_upd + (k * 16 + m) * 12;
      v8h pv = *(const v8h*)(s_pval + k * 8192 + m * 512 + c * 8);
      v8h p0 = *(const v8h*)(s_inps + m * 1536 + 512 + c * 8);
      v8h p1 = *(const v8h*)(s_inps + m * 1536 + 1024 + c * 8);
      float val[8];
#pragma unroll
      for (int e = 0; e < 8; e++)
        val[e] = (float)pv[e] + (float)p0[e] * cf[30] + (float)p1[e] * cf[31];
      _Float16* st = s_stacks + ((k * 16 + m) * 4) * 512 + c * 8;
#pragma unroll
      for (int d = 0; d < 4; d++) {
        v8h s = *(const v8h*)(st + d * 512);
        v8h o;
#pragma unroll
        for (int e = 0; e < 8; e++)
          o[e] = (_Float16)((float)s[e] * U[d * 3] + U[d * 3 + 1] +
                            val[e] * U[d * 3 + 2]);
        *(v8h*)(st + d * 512) = o;
      }
    }
    if (tid < 128) s_ptr[tid] = s_ptr2[tid];
    __syncthreads();

    // ---- step 6: peeks2 + norms/A-fragments for vecs 1,2 ----
    for (int i = tid; i < 2 * 16 * 64; i += 256) {
      int c = i & 63, m = (i >> 6) & 15, k = i >> 10;
      const float* pp = s_ptr + (k * 16 + m) * 4;
      const v8h* st = (const v8h*)(s_stacks + ((k * 16 + m) * 4) * 512) + c;
      v8h s0 = st[0], s1 = st[64], s2 = st[128], s3 = st[192];
      v8h o;
#pragma unroll
      for (int e = 0; e < 8; e++)
        o[e] = (_Float16)((float)s0[e] * pp[0] + (float)s1[e] * pp[1] +
                          (float)s2[e] * pp[2] + (float)s3[e] * pp[3]);
      *(v8h*)(s_inps + m * 1536 + 512 + k * 512 + c * 8) = o;
    }
    __syncthreads();
    if (tid < 128) {  // 32 (m,j) groups x 4 partials (j = 1,2)
      int g2 = tid >> 2, part = tid & 3;
      int m = g2 & 15, j = 1 + (g2 >> 4);
      const v8h* q = (const v8h*)(s_inps + m * 1536 + j * 512 + part * 128);
      float ss = 0.f;
#pragma unroll
      for (int c = 0; c < 16; c++) {
        v8h h = q[c];
#pragma unroll
        for (int e = 0; e < 8; e++) { float f = (float)h[e]; ss += f * f; }
      }
      s_nacc4[(j * 16 + m) * 4 + part] = ss;
    }
    __syncthreads();
    for (int i = tid; i < 2 * 16 * 32; i += 256) {
      int l = i & 31, kt = (i >> 5) & 15, j = 1 + (i >> 9);
      int m = l & 15, kh = l >> 4;
      int g = j * 16 + m;
      float ss = s_nacc4[g * 4] + s_nacc4[g * 4 + 1] +
                 s_nacc4[g * 4 + 2] + s_nacc4[g * 4 + 3];
      float invn = 1.f / fmaxf(sqrtf(ss), 1e-8f);
      const v8h* q = (const v8h*)(s_inps + m * 1536 + j * 512 + kt * 32 + kh * 8);
      v8h c0 = q[0], c1 = q[2];
      v8h o0, o1;
#pragma unroll
      for (int e = 0; e < 8; e++) {
        o0[e] = (_Float16)((float)c0[e] * invn);
        o1[e] = (_Float16)((float)c1[e] * invn);
      }
      v8h* dst = (v8h*)(s_aswz + ((j * 16 + kt) * 32 + l) * 16);
      dst[0] = o0; dst[1] = o1;
    }
    __syncthreads();

    // ---- step 7: calc gates (64 cols) + output mix ----
    for (int nt = wv; nt < 4; nt += 8) {
      float prod[8];
#pragma unroll
      for (int r = 0; r < 8; r++) prod[r] = 1.f;
      int col = nt * 16 + (lane & 15);
#pragma unroll
      for (int j = 0; j < 3; j++) {
        v8f acc = {};
#pragma unroll
        for (int kt = 0; kt < 16; kt++) {
          v16h a = *(const v16h*)(s_aswz + ((j * 16 + kt) * 32 + lane) * 16);
          v16h b = *(const v16h*)(Wswz2 + (((nt * 3 + j) * 16 + kt) * 32 + lane) * 16);
          acc = __builtin_amdgcn_wmma_f32_16x16x32_f16(
              false, a, false, b, (short)0, acc, false, false);
        }
        float nws = nw2[col * 3 + j];
#pragma unroll
        for (int r = 0; r < 8; r++) {
          float c = fabsf(acc[r]);
          prod[r] *= nws * c + (1.f - nws) * (1.f - c);
        }
      }
#pragma unroll
      for (int r = 0; r < 8; r++)
        s_gates[(r + ((lane >> 4) << 3)) * 144 + col] = prod[r];
    }
    __syncthreads();
    for (int i = tid; i < 512; i += 256) {
      int c = i & 31, m = i >> 5;
      s_coef[m * 32 + c] = s_gates[m * 144 + c * 2] + s_gates[m * 144 + c * 2 + 1];
    }
    __syncthreads();
    if (tid < 32) {
      int l = tid, m = l & 15, kh = l >> 4;
      const float* cf = s_coef + m * 32;
      v8h o0, o1;
#pragma unroll
      for (int e = 0; e < 8; e++) {
        o0[e] = (_Float16)cf[kh * 8 + e];
        o1[e] = (_Float16)cf[16 + kh * 8 + e];
      }
      v8h* dst = (v8h*)(s_cswz + l * 16);
      dst[0] = o0; dst[1] = o1;
    }
    __syncthreads();
    for (int vt = wv; vt < 32; vt += 8) {
      v8f acc = {};
      v16h a = *(const v16h*)(s_cswz + lane * 16);
      v16h b = *(const v16h*)(symsSwz + (vt * 32 + lane) * 16);
      acc = __builtin_amdgcn_wmma_f32_16x16x32_f16(
          false, a, false, b, (short)0, acc, false, false);
      int v = vt * 16 + (lane & 15);
#pragma unroll
      for (int r = 0; r < 8; r++) {
        int m = r + ((lane >> 4) << 3);
        const float* cf = s_coef + m * 32;
        out[((size_t)(m0 + m) * SS + t) * VV + v] =
            acc[r] + (float)s_inps[m * 1536 + 512 + v] * cf[30] +
                     (float)s_inps[m * 1536 + 1024 + v] * cf[31];
      }
    }
    __syncthreads();
  }
}

extern "C" void kernel_launch(void* const* d_in, const int* in_sizes, int n_in,
                              void* d_out, int out_size, void* d_ws, size_t ws_size,
                              hipStream_t stream) {
  const float* x         = (const float*)d_in[0];
  const float* syms      = (const float*)d_in[1];
  const float* siv       = (const float*)d_in[2];
  const float* sharp     = (const float*)d_in[3];
  const float* pop_w     = (const float*)d_in[4];
  const float* pop_nw    = (const float*)d_in[5];
  const float* pushop_w  = (const float*)d_in[6];
  const float* pushop_nw = (const float*)d_in[7];
  const float* pushfn_w  = (const float*)d_in[8];
  const float* pushfn_nw = (const float*)d_in[9];
  const float* calc_w    = (const float*)d_in[10];
  const float* calc_nw   = (const float*)d_in[11];
  float* out = (float*)d_out;

  // workspace layout
  char* ws = (char*)d_ws;
  float*    invw    = (float*)(ws + 0);            // 208*3
  float*    nw1     = (float*)(ws + 2496);         // 144*3
  float*    nw2     = (float*)(ws + 4224);         // 64*3
  _Float16* Wswz1   = (_Float16*)(ws + 4992);      // 9*3*16*512
  _Float16* Wswz2   = (_Float16*)(ws + 447360);    // 4*3*16*512
  _Float16* symsSwz = (_Float16*)(ws + 643968);    // 32*512

  prep_norms<<<624, 256, 0, stream>>>(pop_w, pushop_w, pushfn_w, calc_w, invw);
  prep_wswz<<<1248, 256, 0, stream>>>(pop_w, pushop_w, pushfn_w, calc_w,
                                      invw, Wswz1, Wswz2);
  prep_nw<<<3, 256, 0, stream>>>(pop_nw, pushop_nw, pushfn_nw, calc_nw, nw1, nw2);
  prep_syms<<<64, 256, 0, stream>>>(syms, symsSwz);

  const int smem_bytes = 280448;
  (void)hipFuncSetAttribute((const void*)cyborg_main,
                            hipFuncAttributeMaxDynamicSharedMemorySize, smem_bytes);
  cyborg_main<<<dim3(16), dim3(256), smem_bytes, stream>>>(
      x, siv, sharp, nw1, nw2, Wswz1, Wswz2, symsSwz, out);
}
